// CausalEncoder_3006477107613
// MI455X (gfx1250) — matrix operand order repaired and checked by reading, not compiled, not run on hardware
//
#include <hip/hip_runtime.h>
#include <hip/hip_bf16.h>
#include <stdint.h>

// Problem constants (match reference)
#define BATCH   1024
#define NVARS   64
#define NLAG    6
#define D_MODEL 512
#define K_1     1536               // 3*D
#define NROWS   1152               // 64 * 3 * 6 distinct (v, cat, lag) rows

// LDS strides (in halfs), padded for bank-conflict-free column access
#define STRA    1538               // A tile 16 x 1536  (1538*2B rows -> bank step 769, odd)
#define STRW    34                 // W slice transposed 512 x 32  (34*2B rows -> bank step 17)
#define STRH    514                // h tile 16 x 512

typedef __attribute__((ext_vector_type(16))) _Float16 v16h;
typedef __attribute__((ext_vector_type(8)))  float    v8f;

union Frag16 { v16h v; unsigned u[8]; };

// ---------------------------------------------------------------------------
// Kernel 1: build lut[1152][512] = relu(combined @ W1 + b1) @ W2 + b2
// combined row r = [var_table[r/18] | strength_table[(r%18)/6] | lag_table[r%6]]
// 72 blocks x 256 threads (8 waves). Each block: 16 rows x full N=512.
// Wave w owns N columns [w*64, w*64+64) as 4 WMMA n-tiles.
// ---------------------------------------------------------------------------
__global__ __launch_bounds__(256) void causal_lut_kernel(
    const float* __restrict__ var_t, const float* __restrict__ str_t,
    const float* __restrict__ lag_t,
    const float* __restrict__ W1, const float* __restrict__ b1,
    const float* __restrict__ W2, const float* __restrict__ b2,
    float* __restrict__ lut)
{
    __shared__ _Float16 lds_a[16 * STRA];      // combined tile, f16, row-major [m][k]
    __shared__ _Float16 lds_w[D_MODEL * STRW]; // weight k-slice, transposed [n][kk]
    __shared__ _Float16 lds_h[16 * STRH];      // relu(h) tile, f16, [m][n]

    const int t    = threadIdx.x;
    const int lane = t & 31;
    const int wave = t >> 5;          // 0..7
    const int hl   = lane >> 4;       // lane half: 0 = lanes 0-15, 1 = lanes 16-31
    const int l15  = lane & 15;
    const int m0   = blockIdx.x * 16; // row base (NROWS/16 = 72 blocks, exact)

    // ---- stage A tile: 16 rows x 1536 cols, f32 -> f16 (coalesced reads) ----
    for (int m = 0; m < 16; ++m) {
        const int r   = m0 + m;
        const int vi  = r / 18;
        const int ci  = (r % 18) / 6;
        const int li  = r % 6;
        const float* s0 = var_t + vi * D_MODEL;
        const float* s1 = str_t + ci * D_MODEL;
        const float* s2 = lag_t + li * D_MODEL;
        for (int k = t; k < K_1; k += 256) {
            float x = (k < 512) ? s0[k] : (k < 1024) ? s1[k - 512] : s2[k - 1024];
            lds_a[m * STRA + k] = (_Float16)x;
        }
    }

    v8f acc[4];
    const v8f vzero = {0.f, 0.f, 0.f, 0.f, 0.f, 0.f, 0.f, 0.f};
    for (int j = 0; j < 4; ++j) acc[j] = vzero;

    // ================= GEMM1: (16 x 1536) @ W1 (1536 x 512) =================
    for (int kb = 0; kb < K_1 / 32; ++kb) {
        const int k0 = kb * 32;
        __syncthreads();
        // load W1 slice [k0..k0+31] x [0..511] transposed into LDS as f16
        for (int i = t; i < 32 * D_MODEL; i += 256) {
            const int kk = i >> 9;         // 0..31
            const int n  = i & 511;        // 0..511 (coalesced global read)
            lds_w[n * STRW + kk] = (_Float16)W1[(size_t)(k0 + kk) * D_MODEL + n];
        }
        __syncthreads();

        // A fragment (16x32 f16, contiguous k = 16*hl + s): lane l15 = M, hl selects K-half
        Frag16 af;
        {
            const _Float16* ap = lds_a + l15 * STRA + k0 + hl * 16;
            af.u[0] = *(const unsigned*)(ap + 0);
            af.u[1] = *(const unsigned*)(ap + 2);
            af.u[2] = *(const unsigned*)(ap + 4);
            af.u[3] = *(const unsigned*)(ap + 6);
            af.u[4] = *(const unsigned*)(ap + 8);
            af.u[5] = *(const unsigned*)(ap + 10);
            af.u[6] = *(const unsigned*)(ap + 12);
            af.u[7] = *(const unsigned*)(ap + 14);
        }
#pragma unroll
        for (int j = 0; j < 4; ++j) {
            const int n = wave * 64 + j * 16 + l15;
            Frag16 bf;
            const _Float16* bp = lds_w + n * STRW + hl * 16;
            bf.u[0] = *(const unsigned*)(bp + 0);
            bf.u[1] = *(const unsigned*)(bp + 2);
            bf.u[2] = *(const unsigned*)(bp + 4);
            bf.u[3] = *(const unsigned*)(bp + 6);
            bf.u[4] = *(const unsigned*)(bp + 8);
            bf.u[5] = *(const unsigned*)(bp + 10);
            bf.u[6] = *(const unsigned*)(bp + 12);
            bf.u[7] = *(const unsigned*)(bp + 14);
            acc[j] = __builtin_amdgcn_wmma_f32_16x16x32_f16(
                false, af.v, false, bf.v, (short)0, acc[j], false, false);
        }
    }

    // ---- bias + relu, stash h as f16 (C/D layout: VGPR v -> M = v + 8*hl) ----
    __syncthreads();
#pragma unroll
    for (int j = 0; j < 4; ++j) {
        const int n = wave * 64 + j * 16 + l15;
        const float bias = b1[n];
#pragma unroll
        for (int e = 0; e < 8; ++e) {
            float x = acc[j][e] + bias;
            x = x > 0.f ? x : 0.f;
            lds_h[(e + 8 * hl) * STRH + n] = (_Float16)x;
        }
        acc[j] = vzero;
    }

    // ================= GEMM2: (16 x 512) @ W2 (512 x 512) ====================
    for (int kb = 0; kb < D_MODEL / 32; ++kb) {
        const int k0 = kb * 32;
        __syncthreads();
        for (int i = t; i < 32 * D_MODEL; i += 256) {
            const int kk = i >> 9;
            const int n  = i & 511;
            lds_w[n * STRW + kk] = (_Float16)W2[(size_t)(k0 + kk) * D_MODEL + n];
        }
        __syncthreads();

        Frag16 af;
        {
            const _Float16* ap = lds_h + l15 * STRH + k0 + hl * 16;
            af.u[0] = *(const unsigned*)(ap + 0);
            af.u[1] = *(const unsigned*)(ap + 2);
            af.u[2] = *(const unsigned*)(ap + 4);
            af.u[3] = *(const unsigned*)(ap + 6);
            af.u[4] = *(const unsigned*)(ap + 8);
            af.u[5] = *(const unsigned*)(ap + 10);
            af.u[6] = *(const unsigned*)(ap + 12);
            af.u[7] = *(const unsigned*)(ap + 14);
        }
#pragma unroll
        for (int j = 0; j < 4; ++j) {
            const int n = wave * 64 + j * 16 + l15;
            Frag16 bf;
            const _Float16* bp = lds_w + n * STRW + hl * 16;
            bf.u[0] = *(const unsigned*)(bp + 0);
            bf.u[1] = *(const unsigned*)(bp + 2);
            bf.u[2] = *(const unsigned*)(bp + 4);
            bf.u[3] = *(const unsigned*)(bp + 6);
            bf.u[4] = *(const unsigned*)(bp + 8);
            bf.u[5] = *(const unsigned*)(bp + 10);
            bf.u[6] = *(const unsigned*)(bp + 12);
            bf.u[7] = *(const unsigned*)(bp + 14);
            acc[j] = __builtin_amdgcn_wmma_f32_16x16x32_f16(
                false, af.v, false, bf.v, (short)0, acc[j], false, false);
        }
    }

    // ---- bias + store LUT rows (f32) ----
#pragma unroll
    for (int j = 0; j < 4; ++j) {
        const int n = wave * 64 + j * 16 + l15;
        const float bias = b2[n];
#pragma unroll
        for (int e = 0; e < 8; ++e) {
            const int m = e + 8 * hl;
            lut[(size_t)(m0 + m) * D_MODEL + n] = acc[j][e] + bias;
        }
    }
}

// ---------------------------------------------------------------------------
// Kernel 2 (fused): per-(b,v) reduction over causal_matrix -> (cat, lag) ->
// LUT row index, then copy 512-float LUT row to out with float4 stores.
// Block = 256 threads = 4 batches x 64 vars. Grid = B/4 = 256 blocks.
// ---------------------------------------------------------------------------
__global__ __launch_bounds__(256) void causal_encode_kernel(
    const float* __restrict__ cm, const float* __restrict__ lut,
    float* __restrict__ out)
{
    __shared__ int idx_s[256];

    const int t   = threadIdx.x;
    const int sub = t >> 6;           // 0..3
    const int v   = t & 63;
    const int b   = blockIdx.x * 4 + sub;

    // causes[b, v, c, l] = cm[b, c, v, l]; reduce over c (stride 64*6) and l
    float s0 = 0.f, s1 = 0.f, s2 = 0.f, s3 = 0.f, s4 = 0.f, s5 = 0.f, tot = 0.f;
    const float* base = cm + (size_t)b * NVARS * NVARS * NLAG + v * NLAG;
#pragma unroll 4
    for (int c = 0; c < NVARS; ++c) {
        const float* p = base + c * (NVARS * NLAG);     // 8B-aligned (24B steps)
        const float2 x01 = *(const float2*)(p + 0);
        const float2 x23 = *(const float2*)(p + 2);
        const float2 x45 = *(const float2*)(p + 4);
        tot += x01.x + x01.y + x23.x + x23.y + x45.x + x45.y;
        s0 += fabsf(x01.x); s1 += fabsf(x01.y); s2 += fabsf(x23.x);
        s3 += fabsf(x23.y); s4 += fabsf(x45.x); s5 += fabsf(x45.y);
    }
    const float mean = tot * (1.0f / 384.0f);
    const int cat = (mean > 0.1f) ? 1 : ((mean < -0.1f) ? 2 : 0);

    // first-max argmax over lags (strict > matches jnp.argmax tiebreak)
    int lag = 0; float best = s0;
    if (s1 > best) { best = s1; lag = 1; }
    if (s2 > best) { best = s2; lag = 2; }
    if (s3 > best) { best = s3; lag = 3; }
    if (s4 > best) { best = s4; lag = 4; }
    if (s5 > best) { best = s5; lag = 5; }

    idx_s[t] = (v * 3 + cat) * 6 + lag;
    __syncthreads();

    // gather-copy: 256 rows x 128 float4 per block; LUT rows live in L2
    const float4* __restrict__ lut4 = (const float4*)lut;
    float4* __restrict__ out4 = (float4*)out;
    const size_t rowbase = (size_t)blockIdx.x * 256;
#pragma unroll 4
    for (int i = 0; i < 128; ++i) {
        const int linear = t + 256 * i;
        const int row = linear >> 7;      // 0..255
        const int col = linear & 127;     // 0..127
        out4[(rowbase + row) * 128 + col] = lut4[(size_t)idx_s[row] * 128 + col];
    }
}

// ---------------------------------------------------------------------------
extern "C" void kernel_launch(void* const* d_in, const int* in_sizes, int n_in,
                              void* d_out, int out_size, void* d_ws, size_t ws_size,
                              hipStream_t stream) {
    const float* cm    = (const float*)d_in[0];  // (B, V, V, L1)
    const float* var_t = (const float*)d_in[1];  // (V, D)
    const float* str_t = (const float*)d_in[2];  // (3, D)
    const float* lag_t = (const float*)d_in[3];  // (L1, D)
    const float* W1    = (const float*)d_in[4];  // (3D, D)
    const float* b1    = (const float*)d_in[5];  // (D,)
    const float* W2    = (const float*)d_in[6];  // (D, D)
    const float* b2    = (const float*)d_in[7];  // (D,)

    float* lut = (float*)d_ws;                   // NROWS * D floats (~2.25 MB)
    float* out = (float*)d_out;                  // (B, V, D)

    causal_lut_kernel<<<NROWS / 16, 256, 0, stream>>>(
        var_t, str_t, lag_t, W1, b1, W2, b2, lut);
    causal_encode_kernel<<<BATCH / 4, 256, 0, stream>>>(cm, lut, out);
}
